// SpiralMambaBlock_70772471104072
// MI455X (gfx1250) — compile-verified, matrix operand order checked by reading, not run on hardware
//
#include <hip/hip_runtime.h>
#include <hip/hip_bf16.h>
#include <math.h>

// ---------------------------------------------------------------------------
// Types for CDNA5 WMMA / TDM
// ---------------------------------------------------------------------------
typedef __attribute__((ext_vector_type(16))) __bf16 v16bf;
typedef __attribute__((ext_vector_type(8)))  float  v8f;
typedef __attribute__((ext_vector_type(4)))  unsigned int u32x4;
typedef __attribute__((ext_vector_type(8)))  int          i32x8;
typedef __attribute__((ext_vector_type(4)))  int          i32x4;

#if __has_builtin(__builtin_amdgcn_tensor_load_to_lds)
#define HAVE_TDM 1
#else
#define HAVE_TDM 0
#endif

#define AS3 __attribute__((address_space(3)))

// Problem constants (from reference setup_inputs)
#define BB   4
#define HH   32
#define WW2  32
#define LL   1024          // HH*WW2
#define CC   768
#define C2   1536          // 2*CC
#define NST  16            // SSM state dim
#define DR   48            // dt_rank
#define KC   4             // conv kernel
#define BL   (BB*LL)       // 4096 rows

// ---------------------------------------------------------------------------
// helpers
// ---------------------------------------------------------------------------
__device__ __forceinline__ unsigned short f2bf(float f) {
    unsigned int u = __float_as_uint(f);
    unsigned int r = u + 0x7FFFu + ((u >> 16) & 1u);   // round-to-nearest-even
    return (unsigned short)(r >> 16);
}
__device__ __forceinline__ float silu_f(float z) {
    return z / (1.0f + __expf(-z));
}
__device__ __forceinline__ void wait_tensorcnt0() {
#if __has_builtin(__builtin_amdgcn_s_wait_tensorcnt)
    __builtin_amdgcn_s_wait_tensorcnt(0);
#else
    asm volatile("s_wait_tensorcnt 0x0" ::: "memory");
#endif
}

// ---------------------------------------------------------------------------
// 0) spiral index generation (device-side, data independent)
// ---------------------------------------------------------------------------
__global__ void spiral_idx_kernel(int* __restrict__ idx_cw,
                                  int* __restrict__ idx_ccw,
                                  int* __restrict__ rev_cw,
                                  int* __restrict__ rev_ccw) {
    int tid = threadIdx.x;
    if (tid == 0) {
        int top = 0, bottom = HH - 1, left = 0, right = WW2 - 1, p = 0;
        while (top <= bottom && left <= right) {
            for (int j = left; j <= right; ++j) idx_cw[p++] = top * WW2 + j;
            top++;
            for (int i = top; i <= bottom; ++i) idx_cw[p++] = i * WW2 + right;
            right--;
            if (top <= bottom) {
                for (int j = right; j >= left; --j) idx_cw[p++] = bottom * WW2 + j;
                bottom--;
            }
            if (left <= right) {
                for (int i = bottom; i >= top; --i) idx_cw[p++] = i * WW2 + left;
                left++;
            }
        }
    } else if (tid == 1) {
        int top = 0, bottom = HH - 1, left = 0, right = WW2 - 1, p = 0;
        while (top <= bottom && left <= right) {
            for (int j = right; j >= left; --j) idx_ccw[p++] = top * WW2 + j;
            top++;
            for (int i = top; i <= bottom; ++i) idx_ccw[p++] = i * WW2 + left;
            left++;
            if (top <= bottom) {
                for (int j = left; j <= right; ++j) idx_ccw[p++] = bottom * WW2 + j;
                bottom--;
            }
            if (left <= right) {
                for (int i = bottom; i >= top; --i) idx_ccw[p++] = i * WW2 + right;
                right--;
            }
        }
    }
    __syncthreads();
    for (int j = tid; j < LL; j += blockDim.x) {
        rev_cw[idx_cw[j]]   = j;
        rev_ccw[idx_ccw[j]] = j;
    }
}

// ---------------------------------------------------------------------------
// 1) LayerNorm: x (BL, C) f32 -> xn bf16
// ---------------------------------------------------------------------------
__global__ __launch_bounds__(256)
void layernorm_kernel(const float* __restrict__ x,
                      const float* __restrict__ w,
                      const float* __restrict__ b,
                      unsigned short* __restrict__ xn) {
    __shared__ float s_sum[256];
    __shared__ float s_sq[256];
    int row = blockIdx.x;
    int tid = threadIdx.x;
    const float* xr = x + (size_t)row * CC;
    float sum = 0.f, sq = 0.f;
    for (int c = tid; c < CC; c += 256) {
        float v = xr[c];
        sum += v; sq += v * v;
    }
    s_sum[tid] = sum; s_sq[tid] = sq;
    __syncthreads();
    for (int s = 128; s > 0; s >>= 1) {
        if (tid < s) { s_sum[tid] += s_sum[tid + s]; s_sq[tid] += s_sq[tid + s]; }
        __syncthreads();
    }
    float mu  = s_sum[0] / (float)CC;
    float var = s_sq[0] / (float)CC - mu * mu;
    float inv = rsqrtf(var + 1e-6f);
    for (int c = tid; c < CC; c += 256) {
        float v = (xr[c] - mu) * inv * w[c] + b[c];
        xn[(size_t)row * CC + c] = f2bf(v);
    }
}

// ---------------------------------------------------------------------------
// 2) f32 -> bf16 conversion (weights)
// ---------------------------------------------------------------------------
__global__ void cvt_bf16_kernel(const float* __restrict__ src,
                                unsigned short* __restrict__ dst, int n) {
    for (int i = blockIdx.x * blockDim.x + threadIdx.x; i < n;
         i += gridDim.x * blockDim.x)
        dst[i] = f2bf(src[i]);
}

// ---------------------------------------------------------------------------
// 3) WMMA GEMM, block tile 128x128x32, 8 waves (wave tile 32x64).
//    FULL: M%128==0 && N%128==0 && K%32==0 (no guards).
//    FULL && !GATHER: A tile fetched by the Tensor Data Mover into LDS
//    (2-D descriptor, pad 16B per 64B row to reproduce the padded layout).
// ---------------------------------------------------------------------------
#define GBM 128
#define GBN 128
#define GBK 32
#define LDSW 40   // 32 + 8 pad (ushorts); row = 80B, 16B aligned

#if HAVE_TDM
__device__ __forceinline__ void tdm_load_tile_a(const unsigned short* gsrc,
                                                unsigned ldsOff, int lda) {
    unsigned long long ga = (unsigned long long)gsrc;
    u32x4 g0;
    g0[0] = 0x1u;                                       // count=1, user mode
    g0[1] = ldsOff;                                     // lds_addr (bytes)
    g0[2] = (unsigned)(ga & 0xFFFFFFFFull);             // global_addr[31:0]
    g0[3] = (unsigned)((ga >> 32) & 0x01FFFFFFull)      // global_addr[56:32]
          | (2u << 30);                                 // type=2 ("image")
    i32x8 g1;
    g1[0] = (1 << 16)        // data_size = 2 bytes
          | (1 << 20)        // pad_enable
          | (3 << 22)        // pad_interval: 16 DWORDs (64B) before padding
          | (3 << 25);       // pad_amount: 4 DWORDs (16B)
    g1[1] = (32 << 16);      // tensor_dim0 = 32 (lo 16 in bits 31:16)
    g1[2] = (GBM << 16);     // tensor_dim0 hi = 0 ; tensor_dim1 = 128
    g1[3] = (32 << 16);      // tensor_dim1 hi = 0 ; tile_dim0 = 32
    g1[4] = GBM;             // tile_dim1 = 128 ; tile_dim2 = 0
    g1[5] = lda;             // tensor_dim0_stride (elements), low 32
    g1[6] = 0;               // stride hi / dim1_stride lo
    g1[7] = 0;
    i32x4 z4 = (i32x4)(0);
#if defined(__clang_major__) && (__clang_major__ >= 23)
    i32x8 z8 = (i32x8)(0);
    __builtin_amdgcn_tensor_load_to_lds(g0, g1, z4, z4, z8, 0);
#else
    __builtin_amdgcn_tensor_load_to_lds(g0, g1, z4, z4, 0);
#endif
}
#endif

// ACT: 0 = none, 1 = softplus, 2 = gelu(exact)
template <int ACT, bool GATHER, bool FULL>
__global__ __launch_bounds__(256)
void gemm_bf16_kernel(const unsigned short* __restrict__ A, int lda,
                      const unsigned short* __restrict__ Bm, int ldb,
                      const float* __restrict__ bias,
                      const float* __restrict__ residual,
                      float* __restrict__ outF,
                      unsigned short* __restrict__ outB,
                      int M, int N, int K,
                      const int* __restrict__ gather, int Lseq) {
    __shared__ __align__(16) unsigned short As[GBM * LDSW];
    __shared__ __align__(16) unsigned short Bs[GBN * LDSW];

    constexpr bool kTdm = (HAVE_TDM != 0) && FULL && !GATHER;

    int tid  = threadIdx.x;
    int lane = tid & 31;
    int wave = tid >> 5;
    int wm = wave >> 1;       // 0..3  -> 32-row slab
    int wn = wave & 1;        // 0..1  -> 64-col slab
    int blockRow = blockIdx.y * GBM;
    int blockCol = blockIdx.x * GBN;

    v8f acc[2][4];
#pragma unroll
    for (int i = 0; i < 2; ++i)
#pragma unroll
        for (int j = 0; j < 4; ++j) acc[i][j] = (v8f)(0.0f);

    int hk = (lane >> 4) * 8;     // K-half offset per ISA 16-bit frag layout
    int lm = lane & 15;

    // ---- hoist A-row mapping (K-invariant): 2 segments of 8 per thread
    const unsigned short* aSrc[2];
    int  aCs[2];
    int  aR[2];
    bool aOk[2];
    if constexpr (!kTdm) {
#pragma unroll
        for (int s = 0; s < 2; ++s) {
            int seg = tid * 2 + s;          // 0..511
            int r   = seg >> 2;             // 0..127
            int cs  = (seg & 3) * 8;        // 0,8,16,24
            int grow = blockRow + r;
            aR[s] = r; aCs[s] = cs;
            aOk[s] = FULL || (grow < M);
            int grc = aOk[s] ? grow : 0;
            int arow = GATHER ? ((grc / Lseq) * Lseq + gather[grc % Lseq]) : grc;
            aSrc[s] = A + (size_t)arow * lda + cs;
        }
    }
#if HAVE_TDM
    unsigned asOff = 0;
    if constexpr (kTdm)
        asOff = (unsigned)(size_t)(AS3 unsigned short*)&As[0];
#endif

    for (int kt = 0; kt < K; kt += GBK) {
        // ---- A tile -> LDS
        if constexpr (kTdm) {
#if HAVE_TDM
            if (wave == 0) {
                tdm_load_tile_a(A + (size_t)blockRow * lda + kt, asOff, lda);
                wait_tensorcnt0();
            }
#endif
        } else {
#pragma unroll
            for (int s = 0; s < 2; ++s) {
                unsigned short vals[8];
                if (FULL) {
                    *(uint4*)vals = *(const uint4*)(aSrc[s] + kt);
                } else if (aOk[s]) {
                    const unsigned short* src = aSrc[s] + kt;
                    if (kt + aCs[s] + 8 <= K) {
                        *(uint4*)vals = *(const uint4*)src;
                    } else {
#pragma unroll
                        for (int j = 0; j < 8; ++j)
                            vals[j] = (kt + aCs[s] + j < K) ? src[j]
                                                            : (unsigned short)0;
                    }
                } else {
#pragma unroll
                    for (int j = 0; j < 8; ++j) vals[j] = 0;
                }
                *(uint4*)&As[aR[s] * LDSW + aCs[s]] = *(uint4*)vals;
            }
        }
        // ---- B tile -> LDS transposed: Bs[n][k]
#pragma unroll
        for (int s = 0; s < 2; ++s) {
            int seg = tid * 2 + s;          // 0..511
            int k   = seg >> 4;             // 0..31
            int ng  = (seg & 15) * 8;       // 0,8,...,120
            unsigned short vals[8];
            if (FULL) {
                *(uint4*)vals =
                    *(const uint4*)(Bm + (size_t)(kt + k) * ldb + blockCol + ng);
            } else {
                int gk = kt + k;
                if (gk < K) {
                    const unsigned short* src =
                        Bm + (size_t)gk * ldb + blockCol + ng;
                    if (blockCol + ng + 8 <= N) {
                        *(uint4*)vals = *(const uint4*)src;
                    } else {
#pragma unroll
                        for (int j = 0; j < 8; ++j)
                            vals[j] = (blockCol + ng + j < N)
                                          ? src[j] : (unsigned short)0;
                    }
                } else {
#pragma unroll
                    for (int j = 0; j < 8; ++j) vals[j] = 0;
                }
            }
#pragma unroll
            for (int j = 0; j < 8; ++j)
                Bs[(ng + j) * LDSW + k] = vals[j];
        }
        __syncthreads();

        // ---- fragments + WMMA
        union Frag { uint4 q[2]; v16bf v; };
        Frag af[2], bf[4];
#pragma unroll
        for (int fm = 0; fm < 2; ++fm) {
            int r = wm * 32 + fm * 16 + lm;
            af[fm].q[0] = *(const uint4*)&As[r * LDSW + hk];
            af[fm].q[1] = *(const uint4*)&As[r * LDSW + 16 + hk];
        }
#pragma unroll
        for (int fn = 0; fn < 4; ++fn) {
            int c = wn * 64 + fn * 16 + lm;
            bf[fn].q[0] = *(const uint4*)&Bs[c * LDSW + hk];
            bf[fn].q[1] = *(const uint4*)&Bs[c * LDSW + 16 + hk];
        }
#pragma unroll
        for (int fm = 0; fm < 2; ++fm)
#pragma unroll
            for (int fn = 0; fn < 4; ++fn)
                acc[fm][fn] = __builtin_amdgcn_wmma_f32_16x16x32_bf16(
                    false, af[fm].v, false, bf[fn].v,
                    (short)0, acc[fm][fn], false, false);
        __syncthreads();
    }

    // ---- epilogue (C/D layout: VGPR r -> M = r + 8*(lane>=16), N = lane&15)
#pragma unroll
    for (int fm = 0; fm < 2; ++fm) {
#pragma unroll
        for (int fn = 0; fn < 4; ++fn) {
            int rbase = blockRow + wm * 32 + fm * 16 + ((lane >> 4) << 3);
            int col   = blockCol + wn * 64 + fn * 16 + lm;
            if (!FULL && col >= N) continue;
            float bv = bias ? bias[col] : 0.0f;
#pragma unroll
            for (int r = 0; r < 8; ++r) {
                int row = rbase + r;
                if (!FULL && row >= M) continue;
                float v = acc[fm][fn][r] + bv;
                if (ACT == 1) {                         // softplus
                    v = (v > 20.0f) ? v : log1pf(__expf(v));
                } else if (ACT == 2) {                  // exact GELU
                    v = 0.5f * v * (1.0f + erff(v * 0.70710678118654752f));
                }
                size_t o = (size_t)row * N + col;
                if (residual) v += residual[o];
                if (outF) outF[o] = v;
                if (outB) outB[o] = f2bf(v);
            }
        }
    }
}

// ---------------------------------------------------------------------------
// 4) causal depthwise conv (K=4) + SiLU.  u = xz[..., :C]
// ---------------------------------------------------------------------------
__global__ __launch_bounds__(256)
void conv_silu_kernel(const float* __restrict__ xz,
                      const float* __restrict__ conv_w,   // (C, K)
                      const float* __restrict__ conv_b,   // (C,)
                      float* __restrict__ ucf,
                      unsigned short* __restrict__ ucb) {
    int total = BL * CC;
    for (int i = blockIdx.x * blockDim.x + threadIdx.x; i < total;
         i += gridDim.x * blockDim.x) {
        int c   = i % CC;
        int row = i / CC;          // b*L + l
        int l   = row % LL;
        int b   = row / LL;
        float acc = conv_b[c];
#pragma unroll
        for (int k = 0; k < KC; ++k) {
            int lp = l - (KC - 1) + k;
            if (lp >= 0) {
                float uv = xz[((size_t)(b * LL + lp)) * C2 + c];
                acc += conv_w[c * KC + k] * uv;
            }
        }
        acc = silu_f(acc);
        ucf[i] = acc;
        ucb[i] = f2bf(acc);
    }
}

// ---------------------------------------------------------------------------
// 5) selective scan: wave32 = 2 channels x 16 states, shuffle-reduce over N.
// ---------------------------------------------------------------------------
__global__ __launch_bounds__(256)
void scan_kernel(const float* __restrict__ dtf,    // (BL, C)
                 const float* __restrict__ dbl,    // (BL, 80)
                 const float* __restrict__ ucf,    // (BL, C)
                 const float* __restrict__ xz,     // (BL, 2C)  z at [C..2C)
                 const float* __restrict__ A_log,  // (C, N)
                 const float* __restrict__ Dv,     // (C,)
                 unsigned short* __restrict__ yb)  // (BL, C)
{
    int gtid = blockIdx.x * blockDim.x + threadIdx.x;
    int wid  = gtid >> 5;
    int lane = gtid & 31;
    int sub  = lane >> 4;      // which of 2 channels in this wave
    int n    = lane & 15;      // state index
    int dg   = wid * 2 + sub;  // global (b,d) chain
    if (dg >= BB * CC) return;
    int b = dg / CC;
    int d = dg % CC;

    float Acoef = -__expf(A_log[d * NST + n]);
    float Dd    = Dv[d];
    float h = 0.0f;

    size_t rowBase = (size_t)b * LL;
    for (int l = 0; l < LL; ++l) {
        size_t row = rowBase + l;
        float dtv = dtf[row * CC + d];
        float ucv = ucf[row * CC + d];
        float Bv  = dbl[row * (DR + 2 * NST) + DR + n];
        float Cv  = dbl[row * (DR + 2 * NST) + DR + NST + n];
        float dA  = __expf(dtv * Acoef);
        h = h * dA + dtv * Bv * ucv;
        float p = h * Cv;
        p += __shfl_xor(p, 8, 32);
        p += __shfl_xor(p, 4, 32);
        p += __shfl_xor(p, 2, 32);
        p += __shfl_xor(p, 1, 32);
        if (n == 0) {
            float zv = xz[row * C2 + CC + d];
            float y  = (p + Dd * ucv) * silu_f(zv);
            yb[row * CC + d] = f2bf(y);
        }
    }
}

// ---------------------------------------------------------------------------
// 6) fused concat gather: fused[b,i,c] = y_dir[b, rev_dir[i], c]  (bf16 copy)
// ---------------------------------------------------------------------------
__global__ __launch_bounds__(256)
void fuse_gather_kernel(const unsigned short* __restrict__ ycw,
                        const unsigned short* __restrict__ yccw,
                        const int* __restrict__ rev_cw,
                        const int* __restrict__ rev_ccw,
                        unsigned short* __restrict__ fused) {
    int total = BL * C2;
    for (int i = blockIdx.x * blockDim.x + threadIdx.x; i < total;
         i += gridDim.x * blockDim.x) {
        int c   = i % C2;
        int row = i / C2;
        int l   = row % LL;
        int b   = row / LL;
        unsigned short v;
        if (c < CC) v = ycw [((size_t)(b * LL + rev_cw [l])) * CC + c];
        else        v = yccw[((size_t)(b * LL + rev_ccw[l])) * CC + (c - CC)];
        fused[i] = v;
    }
}

// ---------------------------------------------------------------------------
// launcher
// ---------------------------------------------------------------------------
static inline dim3 gemm_grid(int M, int N) {
    return dim3((N + GBN - 1) / GBN, (M + GBM - 1) / GBM);
}

extern "C" void kernel_launch(void* const* d_in, const int* in_sizes, int n_in,
                              void* d_out, int out_size, void* d_ws, size_t ws_size,
                              hipStream_t stream) {
    const float* x       = (const float*)d_in[0];
    const float* norm_w  = (const float*)d_in[1];
    const float* norm_b  = (const float*)d_in[2];
    const float* fus_w1  = (const float*)d_in[21];
    const float* fus_b1  = (const float*)d_in[22];
    const float* fus_w2  = (const float*)d_in[23];
    const float* fus_b2  = (const float*)d_in[24];
    float* out = (float*)d_out;

    // ---- bump allocator on workspace
    char* wp = (char*)d_ws;
    auto alloc = [&](size_t bytes) -> void* {
        void* r = (void*)wp;
        wp += (bytes + 255) & ~(size_t)255;
        return r;
    };
    int* idx_cw   = (int*)alloc(LL * 4);
    int* idx_ccw  = (int*)alloc(LL * 4);
    int* rev_cw   = (int*)alloc(LL * 4);
    int* rev_ccw  = (int*)alloc(LL * 4);
    unsigned short* xn_bf   = (unsigned short*)alloc((size_t)BL * CC * 2);
    unsigned short* w_in_bf = (unsigned short*)alloc((size_t)CC * C2 * 2);
    unsigned short* w_xp_bf = (unsigned short*)alloc((size_t)CC * (DR + 2 * NST) * 2);
    unsigned short* w_dtp_bf= (unsigned short*)alloc((size_t)DR * CC * 2);
    unsigned short* w_out_bf= (unsigned short*)alloc((size_t)CC * CC * 2);
    unsigned short* fw1_bf  = (unsigned short*)alloc((size_t)C2 * CC * 2);
    unsigned short* fw2_bf  = (unsigned short*)alloc((size_t)CC * CC * 2);
    float*          xz_f    = (float*)alloc((size_t)BL * C2 * 4);
    float*          uc_f    = (float*)alloc((size_t)BL * CC * 4);
    unsigned short* uc_bf   = (unsigned short*)alloc((size_t)BL * CC * 2);
    float*          dbl_f   = (float*)alloc((size_t)BL * (DR + 2 * NST) * 4);
    unsigned short* dbl_bf  = (unsigned short*)alloc((size_t)BL * (DR + 2 * NST) * 2);
    float*          dt_f    = (float*)alloc((size_t)BL * CC * 4);
    unsigned short* y_bf    = (unsigned short*)alloc((size_t)BL * CC * 2);
    unsigned short* ycw_bf  = (unsigned short*)alloc((size_t)BL * CC * 2);
    unsigned short* yccw_bf = (unsigned short*)alloc((size_t)BL * CC * 2);
    unsigned short* fused_bf= (unsigned short*)alloc((size_t)BL * C2 * 2);
    unsigned short* h_bf    = (unsigned short*)alloc((size_t)BL * CC * 2);

    // 0) spiral indices
    spiral_idx_kernel<<<1, 256, 0, stream>>>(idx_cw, idx_ccw, rev_cw, rev_ccw);

    // 1) layernorm
    layernorm_kernel<<<BL, 256, 0, stream>>>(x, norm_w, norm_b, xn_bf);

    // 2) fusion weights to bf16 (once)
    cvt_bf16_kernel<<<2048, 256, 0, stream>>>(fus_w1, fw1_bf, C2 * CC);
    cvt_bf16_kernel<<<2048, 256, 0, stream>>>(fus_w2, fw2_bf, CC * CC);

    // per-direction pipeline
    for (int dir = 0; dir < 2; ++dir) {
        int base = 3 + dir * 9;
        const float* in_w    = (const float*)d_in[base + 0];
        const float* conv_w  = (const float*)d_in[base + 1];
        const float* conv_b  = (const float*)d_in[base + 2];
        const float* xproj_w = (const float*)d_in[base + 3];
        const float* dtp_w   = (const float*)d_in[base + 4];
        const float* dtp_b   = (const float*)d_in[base + 5];
        const float* A_log   = (const float*)d_in[base + 6];
        const float* Dvec    = (const float*)d_in[base + 7];
        const float* out_w   = (const float*)d_in[base + 8];
        const int* idx_dir = dir ? idx_ccw : idx_cw;
        unsigned short* ydir_bf = dir ? yccw_bf : ycw_bf;

        cvt_bf16_kernel<<<2048, 256, 0, stream>>>(in_w, w_in_bf, CC * C2);
        cvt_bf16_kernel<<<512, 256, 0, stream>>>(xproj_w, w_xp_bf, CC * (DR + 2 * NST));
        cvt_bf16_kernel<<<512, 256, 0, stream>>>(dtp_w, w_dtp_bf, DR * CC);
        cvt_bf16_kernel<<<2048, 256, 0, stream>>>(out_w, w_out_bf, CC * CC);

        // xz = xn_perm @ in_w   (gathered A rows = spiral permutation)
        gemm_bf16_kernel<0, true, true><<<gemm_grid(BL, C2), 256, 0, stream>>>(
            xn_bf, CC, w_in_bf, C2, nullptr, nullptr,
            xz_f, nullptr, BL, C2, CC, idx_dir, LL);

        // causal conv + silu
        conv_silu_kernel<<<4096, 256, 0, stream>>>(xz_f, conv_w, conv_b, uc_f, uc_bf);

        // dbl = uc @ xproj_w   (f32 for B/C vectors, bf16 for dt_rank slice)
        gemm_bf16_kernel<0, false, false>
            <<<gemm_grid(BL, DR + 2 * NST), 256, 0, stream>>>(
            uc_bf, CC, w_xp_bf, DR + 2 * NST, nullptr, nullptr,
            dbl_f, dbl_bf, BL, DR + 2 * NST, CC, nullptr, LL);

        // dt = softplus(dtc @ dtp_w + dtp_b)   (A = dbl_bf, lda=80, K=48)
        gemm_bf16_kernel<1, false, false><<<gemm_grid(BL, CC), 256, 0, stream>>>(
            dbl_bf, DR + 2 * NST, w_dtp_bf, CC, dtp_b, nullptr,
            dt_f, nullptr, BL, CC, DR, nullptr, LL);

        // selective scan -> y (gated by silu(z) and +D*uc)
        scan_kernel<<<(BB * CC / 2 * 32 + 255) / 256, 256, 0, stream>>>(
            dt_f, dbl_f, uc_f, xz_f, A_log, Dvec, y_bf);

        // y @ out_w  (TDM A path; bf16 output, still in spiral order)
        gemm_bf16_kernel<0, false, true><<<gemm_grid(BL, CC), 256, 0, stream>>>(
            y_bf, CC, w_out_bf, CC, nullptr, nullptr,
            nullptr, ydir_bf, BL, CC, CC, nullptr, LL);
    }

    // fused = concat(un-spiral(ycw), un-spiral(yccw))
    fuse_gather_kernel<<<8192, 256, 0, stream>>>(ycw_bf, yccw_bf, rev_cw, rev_ccw,
                                                 fused_bf);

    // h = gelu(fused @ fus_w1 + fus_b1)   (TDM A path)
    gemm_bf16_kernel<2, false, true><<<gemm_grid(BL, CC), 256, 0, stream>>>(
        fused_bf, C2, fw1_bf, CC, fus_b1, nullptr,
        nullptr, h_bf, BL, CC, C2, nullptr, LL);

    // out = h @ fus_w2 + fus_b2 + x   (TDM A path)
    gemm_bf16_kernel<0, false, true><<<gemm_grid(BL, CC), 256, 0, stream>>>(
        h_bf, CC, fw2_bf, CC, fus_b2, x,
        out, nullptr, BL, CC, CC, nullptr, LL);
}
